// CrossAttentionHead_51419348468340
// MI455X (gfx1250) — compile-verified, matrix operand order checked by reading, not run on hardware
//
#include <hip/hip_runtime.h>

// Cross-attention head for MI455X (gfx1250, wave32, WMMA).
// B=8, L=2048, D=1024. Outputs: context [B,L,D] f32 then attn [B,L,L] f32.
//
// Pipeline:
//   1) f32->f16 convert pass (inputs + weights)          [bandwidth trivial]
//   2) q/k/v projections: pure-f16 WMMA GEMM (v stored transposed)
//   3) scores = q.k^T * 1/sqrt(D) with query-row mask -> attn region of d_out
//   4) row softmax in place, also emits f16 attn copy to workspace
//   5) context = attn_f16 @ v^T  (pure-f16 WMMA GEMM, f32 out)

#define B_ 8
#define L_ 2048
#define D_ 1024
#define M_ (B_ * L_)

typedef __attribute__((ext_vector_type(16))) _Float16 v16h;
typedef __attribute__((ext_vector_type(8)))  _Float16 v8h;
typedef __attribute__((ext_vector_type(4)))  _Float16 v4h;
typedef __attribute__((ext_vector_type(8)))  float    v8f;
typedef __attribute__((ext_vector_type(4)))  float    v4f;

// ---- fragment loader --------------------------------------------------------
// A-frag (16x32, lane&15 = row, lane>>4 = K half) and B-frag (32x16, lane&15 =
// col of B = row of the row-major [N,K] operand). A and B use the SAME per-lane
// K striping, so any K permutation cancels in the dot product.
__device__ __forceinline__ v16h frag_f16(const _Float16* __restrict__ base,
                                         int ld, int row0, int k0) {
  const int lane = (int)(threadIdx.x & 31u);
  const _Float16* p = base + (size_t)(row0 + (lane & 15)) * ld
                           + (k0 + ((lane >> 4) << 4));
  return *(const v16h*)p;  // 32B, 32B-aligned -> 2x global_load_b128
}

__device__ __forceinline__ void prefetch_frag(const _Float16* __restrict__ base,
                                              int ld, int row0, int k0) {
  const int lane = (int)(threadIdx.x & 31u);
  const _Float16* p = base + (size_t)(row0 + (lane & 15)) * ld
                           + (k0 + ((lane >> 4) << 4));
  __builtin_prefetch((const void*)p, 0, 3);  // global_prefetch_b8
}

#define WMMA_F16(A, Bb, C) \
  __builtin_amdgcn_wmma_f32_16x16x32_f16(false, (A), false, (Bb), (short)0, (C), false, false)

// ---- 1) elementwise f32 -> f16 ---------------------------------------------
__global__ __launch_bounds__(256) void ca_cvt_f16_kernel(
    const float* __restrict__ in, _Float16* __restrict__ out, int n) {
  const int idx = ((int)blockIdx.x * 256 + (int)threadIdx.x) * 8;
  if (idx >= n) return;
  v4f a = *(const v4f*)(in + idx);
  v4f b = *(const v4f*)(in + idx + 4);
  v8h o;
#pragma unroll
  for (int i = 0; i < 4; ++i) {
    o[i]     = (_Float16)a[i];
    o[4 + i] = (_Float16)b[i];
  }
  *(v8h*)(out + idx) = o;  // 16B store
}

// ---- 2) projection: Y[m,n] = sum_k X[m,k] * W[n,k]  (y = x @ W^T) ----------
// transposed==1: store Y^T per batch into YT[b][n][l] with packed v8h stores.
__global__ __launch_bounds__(128) void ca_proj_kernel(
    const _Float16* __restrict__ X, const _Float16* __restrict__ W,
    _Float16* __restrict__ Y, _Float16* __restrict__ YT, int transposed) {
  const int wave = (int)(threadIdx.x >> 5u);
  const int m0 = (int)blockIdx.x * 128 + (wave & 1) * 64;
  const int n0 = (int)blockIdx.y * 128 + (wave >> 1) * 64;

  v8f acc[4][4];
  v8f zero = {};
#pragma unroll
  for (int i = 0; i < 4; ++i)
#pragma unroll
    for (int j = 0; j < 4; ++j) acc[i][j] = zero;

  for (int k0 = 0; k0 < D_; k0 += 32) {
    v16h a[4], b[4];
#pragma unroll
    for (int i = 0; i < 4; ++i) a[i] = frag_f16(X, D_, m0 + 16 * i, k0);
#pragma unroll
    for (int j = 0; j < 4; ++j) b[j] = frag_f16(W, D_, n0 + 16 * j, k0);
    // branch-free prefetch two K-steps ahead (clamped; always emitted)
    {
      const int kp = (k0 + 64 < D_) ? (k0 + 64) : (D_ - 32);
#pragma unroll
      for (int i = 0; i < 4; ++i) prefetch_frag(X, D_, m0 + 16 * i, kp);
#pragma unroll
      for (int j = 0; j < 4; ++j) prefetch_frag(W, D_, n0 + 16 * j, kp);
    }
#pragma unroll
    for (int i = 0; i < 4; ++i)
#pragma unroll
      for (int j = 0; j < 4; ++j) acc[i][j] = WMMA_F16(a[i], b[j], acc[i][j]);
  }

  const int lane  = (int)(threadIdx.x & 31u);
  const int cn    = lane & 15;
  const int chalf = lane >> 4;
  if (!transposed) {
#pragma unroll
    for (int i = 0; i < 4; ++i)
#pragma unroll
      for (int j = 0; j < 4; ++j) {
        const int col = n0 + 16 * j + cn;
#pragma unroll
        for (int r = 0; r < 8; ++r) {
          const int row = m0 + 16 * i + 8 * chalf + r;
          Y[(size_t)row * D_ + col] = (_Float16)acc[i][j][r];
        }
      }
  } else {
#pragma unroll
    for (int i = 0; i < 4; ++i)
#pragma unroll
      for (int j = 0; j < 4; ++j) {
        const int e     = n0 + 16 * j + cn;
        const int mbase = m0 + 16 * i + 8 * chalf;    // 8 consecutive rows
        const int bi    = mbase >> 11;                // / L_
        const int li    = mbase & (L_ - 1);
        v8h pk;
#pragma unroll
        for (int r = 0; r < 8; ++r) pk[r] = (_Float16)acc[i][j][r];
        *(v8h*)(YT + ((size_t)bi * D_ + e) * L_ + li) = pk;  // 16B packed store
      }
  }
}

// ---- 3) scores: S[b,lq,lk] = (q . k) / sqrt(D), query-row mask --------------
__global__ __launch_bounds__(128) void ca_scores_kernel(
    const _Float16* __restrict__ Q, const _Float16* __restrict__ Kmat,
    const int* __restrict__ lens, float* __restrict__ S) {
  const int bz = (int)blockIdx.z;
  const _Float16* q = Q    + (size_t)bz * L_ * D_;
  const _Float16* k = Kmat + (size_t)bz * L_ * D_;
  float* s = S + (size_t)bz * L_ * L_;

  const int wave = (int)(threadIdx.x >> 5u);
  const int m0 = (int)blockIdx.x * 128 + (wave & 1) * 64;
  const int n0 = (int)blockIdx.y * 128 + (wave >> 1) * 64;

  v8f acc[4][4];
  v8f zero = {};
#pragma unroll
  for (int i = 0; i < 4; ++i)
#pragma unroll
    for (int j = 0; j < 4; ++j) acc[i][j] = zero;

  for (int k0 = 0; k0 < D_; k0 += 32) {
    v16h a[4], b[4];
#pragma unroll
    for (int i = 0; i < 4; ++i) a[i] = frag_f16(q, D_, m0 + 16 * i, k0);
#pragma unroll
    for (int j = 0; j < 4; ++j) b[j] = frag_f16(k, D_, n0 + 16 * j, k0);
    {
      const int kp = (k0 + 64 < D_) ? (k0 + 64) : (D_ - 32);
#pragma unroll
      for (int i = 0; i < 4; ++i) prefetch_frag(q, D_, m0 + 16 * i, kp);
#pragma unroll
      for (int j = 0; j < 4; ++j) prefetch_frag(k, D_, n0 + 16 * j, kp);
    }
#pragma unroll
    for (int i = 0; i < 4; ++i)
#pragma unroll
      for (int j = 0; j < 4; ++j) acc[i][j] = WMMA_F16(a[i], b[j], acc[i][j]);
  }

  const int len   = lens[bz];
  const int lane  = (int)(threadIdx.x & 31u);
  const int cn    = lane & 15;
  const int chalf = lane >> 4;
  const float scale = 0.03125f;  // 1/sqrt(1024)
#pragma unroll
  for (int i = 0; i < 4; ++i)
#pragma unroll
    for (int j = 0; j < 4; ++j) {
      const int col = n0 + 16 * j + cn;
#pragma unroll
      for (int r = 0; r < 8; ++r) {
        const int row = m0 + 16 * i + 8 * chalf + r;  // query index lq
        const float v = (row < len) ? acc[i][j][r] * scale : -1.0e9f;
        s[(size_t)row * L_ + col] = v;
      }
    }
}

// ---- 4) row softmax in place; also emit f16 copy for the context GEMM -------
__global__ __launch_bounds__(256) void ca_softmax_kernel(
    float* __restrict__ S, _Float16* __restrict__ Sh) {
  float*    p  = S  + (size_t)blockIdx.x * L_;
  _Float16* ph = Sh + (size_t)blockIdx.x * L_;
  const int t = (int)threadIdx.x;

  v4f x0 = *(v4f*)(p + 4 * t);
  v4f x1 = *(v4f*)(p + 1024 + 4 * t);

  float mx = x0[0];
#pragma unroll
  for (int i = 1; i < 4; ++i) mx = fmaxf(mx, x0[i]);
#pragma unroll
  for (int i = 0; i < 4; ++i) mx = fmaxf(mx, x1[i]);

  __shared__ float red[256];
  red[t] = mx;
  __syncthreads();
  for (int s = 128; s > 0; s >>= 1) {
    if (t < s) red[t] = fmaxf(red[t], red[t + s]);
    __syncthreads();
  }
  mx = red[0];
  __syncthreads();

  float sum = 0.0f;
#pragma unroll
  for (int i = 0; i < 4; ++i) { x0[i] = __expf(x0[i] - mx); sum += x0[i]; }
#pragma unroll
  for (int i = 0; i < 4; ++i) { x1[i] = __expf(x1[i] - mx); sum += x1[i]; }

  red[t] = sum;
  __syncthreads();
  for (int s = 128; s > 0; s >>= 1) {
    if (t < s) red[t] += red[t + s];
    __syncthreads();
  }
  const float inv = 1.0f / red[0];

  x0 = x0 * inv;
  x1 = x1 * inv;
  *(v4f*)(p + 4 * t)        = x0;
  *(v4f*)(p + 1024 + 4 * t) = x1;

  v4h h0, h1;
#pragma unroll
  for (int i = 0; i < 4; ++i) { h0[i] = (_Float16)x0[i]; h1[i] = (_Float16)x1[i]; }
  *(v4h*)(ph + 4 * t)        = h0;   // 8B packed f16 store
  *(v4h*)(ph + 1024 + 4 * t) = h1;
}

// ---- 5) context: C[b,lq,e] = sum_lk attn[b,lq,lk] * v[b,lk,e] (v stored ^T) -
__global__ __launch_bounds__(128) void ca_context_kernel(
    const _Float16* __restrict__ AttnH, const _Float16* __restrict__ VT,
    float* __restrict__ Cout) {
  const int bz = (int)blockIdx.z;
  const _Float16* attn = AttnH + (size_t)bz * L_ * L_;
  const _Float16* vt   = VT    + (size_t)bz * D_ * L_;
  float* out           = Cout  + (size_t)bz * L_ * D_;

  const int wave = (int)(threadIdx.x >> 5u);
  const int m0 = (int)blockIdx.x * 128 + (wave & 1) * 64;   // query rows
  const int n0 = (int)blockIdx.y * 128 + (wave >> 1) * 64;  // feature cols

  v8f acc[4][4];
  v8f zero = {};
#pragma unroll
  for (int i = 0; i < 4; ++i)
#pragma unroll
    for (int j = 0; j < 4; ++j) acc[i][j] = zero;

  for (int k0 = 0; k0 < L_; k0 += 32) {
    v16h a[4], b[4];
#pragma unroll
    for (int i = 0; i < 4; ++i) a[i] = frag_f16(attn, L_, m0 + 16 * i, k0);
#pragma unroll
    for (int j = 0; j < 4; ++j) b[j] = frag_f16(vt, L_, n0 + 16 * j, k0);
    {
      const int kp = (k0 + 64 < L_) ? (k0 + 64) : (L_ - 32);
#pragma unroll
      for (int i = 0; i < 4; ++i) prefetch_frag(attn, L_, m0 + 16 * i, kp);
#pragma unroll
      for (int j = 0; j < 4; ++j) prefetch_frag(vt, L_, n0 + 16 * j, kp);
    }
#pragma unroll
    for (int i = 0; i < 4; ++i)
#pragma unroll
      for (int j = 0; j < 4; ++j) acc[i][j] = WMMA_F16(a[i], b[j], acc[i][j]);
  }

  const int lane  = (int)(threadIdx.x & 31u);
  const int cn    = lane & 15;
  const int chalf = lane >> 4;
#pragma unroll
  for (int i = 0; i < 4; ++i)
#pragma unroll
    for (int j = 0; j < 4; ++j) {
      const int col = n0 + 16 * j + cn;
#pragma unroll
      for (int r = 0; r < 8; ++r) {
        const int row = m0 + 16 * i + 8 * chalf + r;
        out[(size_t)row * D_ + col] = acc[i][j][r];
      }
    }
}

// ---- host launcher ----------------------------------------------------------
extern "C" void kernel_launch(void* const* d_in, const int* in_sizes, int n_in,
                              void* d_out, int out_size, void* d_ws, size_t ws_size,
                              hipStream_t stream) {
  (void)in_sizes; (void)n_in; (void)out_size; (void)ws_size;

  const float* input_seq  = (const float*)d_in[0];
  const int*   input_lens = (const int*)d_in[1];
  const float* output_seq = (const float*)d_in[2];
  const float* Wk         = (const float*)d_in[3];
  const float* Wv         = (const float*)d_in[4];
  const float* Wq         = (const float*)d_in[5];

  float* ctx  = (float*)d_out;                        // [B,L,D]
  float* attn = (float*)d_out + (size_t)B_ * L_ * D_; // [B,L,L]

  // f16 workspace (~166 MB):
  //   qh, kh, vt : live for whole pipeline           (3 * 32 MB)
  //   xh, yh     : f16 inputs, dead after projections (2 * 32 MB)
  //   attn_h     : overlays xh+yh (B*L*L == 2*M*D elements exactly) (64 MB)
  //   wkh/wvh/wqh: f16 weights                        (3 * 2 MB)
  _Float16* qh    = (_Float16*)d_ws;
  _Float16* kh    = qh + (size_t)M_ * D_;
  _Float16* vt    = kh + (size_t)M_ * D_;
  _Float16* xh    = vt + (size_t)M_ * D_;
  _Float16* yh    = xh + (size_t)M_ * D_;
  _Float16* attnh = xh;                          // overlay
  _Float16* wkh   = yh + (size_t)M_ * D_;
  _Float16* wvh   = wkh + (size_t)D_ * D_;
  _Float16* wqh   = wvh + (size_t)D_ * D_;

  // 1) convert inputs + weights to f16
  ca_cvt_f16_kernel<<<dim3((M_ * D_) / 2048), dim3(256), 0, stream>>>(input_seq,  xh, M_ * D_);
  ca_cvt_f16_kernel<<<dim3((M_ * D_) / 2048), dim3(256), 0, stream>>>(output_seq, yh, M_ * D_);
  ca_cvt_f16_kernel<<<dim3((D_ * D_) / 2048), dim3(256), 0, stream>>>(Wk, wkh, D_ * D_);
  ca_cvt_f16_kernel<<<dim3((D_ * D_) / 2048), dim3(256), 0, stream>>>(Wv, wvh, D_ * D_);
  ca_cvt_f16_kernel<<<dim3((D_ * D_) / 2048), dim3(256), 0, stream>>>(Wq, wqh, D_ * D_);

  // 2) projections (q, k normal layout; v transposed per batch)
  const dim3 blk(128);
  const dim3 gproj(M_ / 128, D_ / 128);  // 128 x 8
  ca_proj_kernel<<<gproj, blk, 0, stream>>>(yh, wqh, qh, nullptr, 0);
  ca_proj_kernel<<<gproj, blk, 0, stream>>>(xh, wkh, kh, nullptr, 0);
  ca_proj_kernel<<<gproj, blk, 0, stream>>>(xh, wvh, nullptr, vt, 1);

  // 3) scores -> attn region of d_out (masked + scaled)
  ca_scores_kernel<<<dim3(L_ / 128, L_ / 128, B_), blk, 0, stream>>>(qh, kh, input_lens, attn);

  // 4) softmax in place + f16 copy (xh/yh are dead now; attnh overlays them)
  ca_softmax_kernel<<<dim3(B_ * L_), dim3(256), 0, stream>>>(attn, attnh);

  // 5) context GEMM
  ca_context_kernel<<<dim3(L_ / 128, D_ / 128, B_), blk, 0, stream>>>(attnh, vt, ctx);
}
